// Attention_1486058684451
// MI455X (gfx1250) — matrix operand order repaired and checked by reading, not compiled, not run on hardware
//
#include <hip/hip_runtime.h>
#include <hip/hip_bf16.h>
#include <math.h>

// ---------------- problem constants ----------------
#define BSZ    8
#define SEQLEN 16
#define DIM    4096
#define NH     32
#define NKV    8
#define HD     128
#define CL     4096
#define KVROW  (NKV*HD)       // 1024
#define QROW   (NH*HD)        // 4096
#define MROWS  (BSZ*SEQLEN)   // 128
#define FA_WAVES 8            // waves per flash-attention block
#define FA_KEYS  (CL/FA_WAVES) // 512 keys per wave

typedef __attribute__((ext_vector_type(16))) __bf16 v16bf;
typedef __attribute__((ext_vector_type(8)))  float  v8f;

static __device__ __forceinline__ v8f wmma_bf16(v16bf a, v16bf b, v8f c) {
  // D = A(16x32 bf16) x B(32x16 bf16) + C(16x16 f32)
  return __builtin_amdgcn_wmma_f32_16x16x32_bf16(false, a, false, b, (short)0, c,
                                                 false, false);
}

// ---------------------------------------------------------------------------
// Kernel 1: KV-cache shift.  cache[:, SEQLEN:] -> out_cache[:, :CL-SEQLEN]
// Per batch one contiguous block -> pure b128 streaming copy.
// ---------------------------------------------------------------------------
__global__ __launch_bounds__(256) void shift_cache_kernel(
    const float* __restrict__ sk, const float* __restrict__ sv,
    float* __restrict__ dk, float* __restrict__ dv) {
  const size_t per_b = (size_t)(CL - SEQLEN) * KVROW / 4; // float4 per batch
  const size_t cstr  = (size_t)CL * KVROW / 4;
  const size_t shift = (size_t)SEQLEN * KVROW / 4;
  const float4* s4k = (const float4*)sk;
  const float4* s4v = (const float4*)sv;
  float4* d4k = (float4*)dk;
  float4* d4v = (float4*)dv;
  const size_t total = (size_t)BSZ * per_b;
  for (size_t i = (size_t)blockIdx.x * blockDim.x + threadIdx.x; i < total;
       i += (size_t)gridDim.x * blockDim.x) {
    size_t b = i / per_b;
    size_t r = i - b * per_b;
    d4k[b * cstr + r] = s4k[b * cstr + shift + r];
    d4v[b * cstr + r] = s4v[b * cstr + shift + r];
  }
}

// ---------------------------------------------------------------------------
// GEMM:  C[M,N] = A[M,K] @ Bw[N,K]^T   (f32 in/out, bf16 WMMA compute)
// One wave per 16x64 C strip: A fragment loaded once per k-step and reused
// across 4 accumulators (4x fewer A loads + conversions per WMMA).
// Fragment layouts used by this code (contiguous k = 16*half + e):
//   A (16x32):  lanes 0-15: M=lane,  K = 0..15
//               lanes16-31: M=lane-16, K = 16..31
//   B (32x16):  lanes 0-15: N=lane,  K = 0..15 ; lanes16-31: N=lane-16, K=16..31
//   C (16x16):  element r -> row r (lanes 0-15) / r+8 (lanes 16-31), col=lane%16
// ---------------------------------------------------------------------------
__global__ __launch_bounds__(256) void gemm_bf16_kernel(
    const float* __restrict__ A, const float* __restrict__ Bw,
    float* __restrict__ C, int M, int N, int K) {
  const int lane = threadIdx.x & 31;
  const int wave = threadIdx.x >> 5;
  const int strip = blockIdx.x * (blockDim.x >> 5) + wave;
  const int nsg  = N >> 6;                  // 64-wide N strips
  const int ntot = (M >> 4) * nsg;
  if (strip >= ntot) return;                // wave-uniform
  const int mt  = strip / nsg;
  const int ng  = strip - mt * nsg;
  const int half = lane >> 4;               // 0 or 1
  const int l16  = lane & 15;
  const int ka0  = half ? 16 : 0;
  const int kb0  = half ? 16 : 0;

  const float* ap = A + (size_t)((mt << 4) + l16) * K;
  const float* bp[4];
#pragma unroll
  for (int j = 0; j < 4; ++j)
    bp[j] = Bw + (size_t)((ng << 6) + (j << 4) + l16) * K;

  v8f acc[4];
#pragma unroll
  for (int j = 0; j < 4; ++j) acc[j] = (v8f){0.f,0.f,0.f,0.f,0.f,0.f,0.f,0.f};

  for (int k = 0; k < K; k += 32) {
    v16bf a;
#pragma unroll
    for (int e = 0; e < 8; ++e) {
      a[e]     = (__bf16)ap[k + ka0 + e];
      a[e + 8] = (__bf16)ap[k + 8 + ka0 + e];
    }
#pragma unroll
    for (int j = 0; j < 4; ++j) {
      // keep the HBM weight stream deep (global_prefetch_b8, ~1KB ahead)
      __builtin_prefetch(bp[j] + k + 256, 0, 1);
      v16bf b;
#pragma unroll
      for (int e = 0; e < 16; ++e) b[e] = (__bf16)bp[j][k + kb0 + e];
      acc[j] = wmma_bf16(a, b, acc[j]);
    }
  }
#pragma unroll
  for (int j = 0; j < 4; ++j) {
    float* cp = C + (size_t)((mt << 4) + half * 8) * N + (ng << 6) + (j << 4) + l16;
#pragma unroll
    for (int r = 0; r < 8; ++r) cp[(size_t)r * N] = acc[j][r];
  }
}

// ---------------------------------------------------------------------------
// Kernel 3: RoPE on xq (in place) and xk (scattered to cache-K tail),
// plus xv -> cache-V tail.
// ---------------------------------------------------------------------------
__global__ __launch_bounds__(256) void rope_scatter_kernel(
    float* __restrict__ xq, const float* __restrict__ xk,
    const float* __restrict__ xv, const float* __restrict__ fc,
    float* __restrict__ ck_out, float* __restrict__ cv_out) {
  const int QP = MROWS * (QROW / 2);   // 262144 q pairs
  const int KP = MROWS * (KVROW / 2);  // 65536 k pairs
  const int VE = MROWS * KVROW;        // 131072 v elems
  const int idx = blockIdx.x * blockDim.x + threadIdx.x;
  if (idx < QP) {
    int m = idx / (QROW / 2), p = idx - m * (QROW / 2);
    int s = m & (SEQLEN - 1), i = p & (HD / 2 - 1);
    float c  = fc[(s * (HD / 2) + i) * 2 + 0];
    float sn = fc[(s * (HD / 2) + i) * 2 + 1];
    size_t o = (size_t)m * QROW + 2 * p;
    float re = xq[o], im = xq[o + 1];
    xq[o]     = re * c - im * sn;
    xq[o + 1] = re * sn + im * c;
  } else if (idx < QP + KP) {
    int j = idx - QP;
    int m = j / (KVROW / 2), p = j - m * (KVROW / 2);
    int bq = m >> 4, s = m & 15, i = p & (HD / 2 - 1);
    float c  = fc[(s * (HD / 2) + i) * 2 + 0];
    float sn = fc[(s * (HD / 2) + i) * 2 + 1];
    float re = xk[(size_t)m * KVROW + 2 * p];
    float im = xk[(size_t)m * KVROW + 2 * p + 1];
    size_t dst = ((size_t)bq * CL + (CL - SEQLEN + s)) * KVROW + 2 * p;
    ck_out[dst]     = re * c - im * sn;
    ck_out[dst + 1] = re * sn + im * c;
  } else if (idx < QP + KP + VE) {
    int j = idx - QP - KP;
    int m = j / KVROW, col = j - m * KVROW;
    int bq = m >> 4, s = m & 15;
    cv_out[((size_t)bq * CL + (CL - SEQLEN + s)) * KVROW + col] =
        xv[(size_t)m * KVROW + col];
  }
}

// ---------------------------------------------------------------------------
// Kernel 4: GQA flash attention.  1 block = (b,h); FA_WAVES waves, each owns
// FA_KEYS keys with online softmax; cross-wave combine via LDS at the end.
// ---------------------------------------------------------------------------
__global__ __launch_bounds__(FA_WAVES * 32) void flash_attn_kernel(
    const float* __restrict__ xq, const float* __restrict__ ck,
    const float* __restrict__ cv, const float* __restrict__ mask,
    float* __restrict__ attn) {
  __shared__ __bf16 p_lds[FA_WAVES][16 * 32];   // P tile staging (C->A transpose)
  __shared__ float  o_lds[FA_WAVES][16 * HD];   // per-wave partial O
  __shared__ float  m_lds[FA_WAVES][16];
  __shared__ float  l_lds[FA_WAVES][16];

  const int lane = threadIdx.x & 31;
  const int w    = threadIdx.x >> 5;
  const int b    = blockIdx.x >> 5;
  const int h    = blockIdx.x & 31;
  const int kvh  = h >> 2;                 // n_rep = 4
  const int half = lane >> 4;
  const int l16  = lane & 15;
  const int ka0  = half ? 16 : 0;
  const int kb0  = half ? 16 : 0;
  const float sc = 0.08838834764831845f;   // 1/sqrt(128)

  // Q A-fragments (4 chunks of K=32 covering head_dim 128)
  const float* qp = xq + (size_t)(b * SEQLEN + l16) * QROW + h * HD;
  v16bf aQ[4];
#pragma unroll
  for (int dc = 0; dc < 4; ++dc) {
#pragma unroll
    for (int e = 0; e < 8; ++e) {
      aQ[dc][e]     = (__bf16)qp[dc * 32 + ka0 + e];
      aQ[dc][e + 8] = (__bf16)qp[dc * 32 + 8 + ka0 + e];
    }
  }

  v8f acc[8];
#pragma unroll
  for (int t = 0; t < 8; ++t) acc[t] = (v8f){0.f,0.f,0.f,0.f,0.f,0.f,0.f,0.f};
  float m_r[8], l_r[8];
#pragma unroll
  for (int r = 0; r < 8; ++r) { m_r[r] = -INFINITY; l_r[r] = 0.f; }

  const size_t kvbase = (size_t)b * CL * KVROW + (size_t)kvh * HD;

  for (int kb = w * FA_KEYS; kb < w * FA_KEYS + FA_KEYS; kb += 32) {
    // ---- scores: two 16x16 tiles (keys kb..kb+15, kb+16..kb+31) ----
    v8f s0 = {0.f,0.f,0.f,0.f,0.f,0.f,0.f,0.f};
    v8f s1 = s0;
    const float* kp0 = ck + kvbase + (size_t)(kb + l16) * KVROW;
    const float* kp1 = ck + kvbase + (size_t)(kb + 16 + l16) * KVROW;
#pragma unroll
    for (int dc = 0; dc < 4; ++dc) {
      v16bf b0, b1;
#pragma unroll
      for (int e = 0; e < 16; ++e) {
        b0[e] = (__bf16)kp0[dc * 32 + kb0 + e];
        b1[e] = (__bf16)kp1[dc * 32 + kb0 + e];
      }
      s0 = wmma_bf16(aQ[dc], b0, s0);
      s1 = wmma_bf16(aQ[dc], b1, s1);
    }
    // ---- online softmax (rows live in 16-lane groups) ----
#pragma unroll
    for (int r = 0; r < 8; ++r) {
      const int rowq = r + half * 8;
      float v0 = s0[r] * sc + mask[rowq * CL + kb + l16];
      float v1 = s1[r] * sc + mask[rowq * CL + kb + 16 + l16];
      float mx = fmaxf(v0, v1);
#pragma unroll
      for (int off = 1; off < 16; off <<= 1)
        mx = fmaxf(mx, __shfl_xor(mx, off, 32));
      float mn   = fmaxf(m_r[r], mx);
      float corr = __expf(m_r[r] - mn);
      float p0   = __expf(v0 - mn);
      float p1   = __expf(v1 - mn);
      float rs   = p0 + p1;
#pragma unroll
      for (int off = 1; off < 16; off <<= 1)
        rs += __shfl_xor(rs, off, 32);
      l_r[r] = l_r[r] * corr + rs;
      m_r[r] = mn;
#pragma unroll
      for (int ct = 0; ct < 8; ++ct) acc[ct][r] *= corr;
      p_lds[w][rowq * 32 + l16]      = (__bf16)p0;
      p_lds[w][rowq * 32 + 16 + l16] = (__bf16)p1;
    }
    // wave-local LDS write->read across lane halves: drain DS counter
    asm volatile("s_wait_dscnt 0x0" ::: "memory");
    // ---- P A-fragment from LDS (row-major 16x32 bf16) ----
    v16bf aP;
    {
      const __bf16* pl = &p_lds[w][l16 * 32];
#pragma unroll
      for (int e = 0; e < 8; ++e) {
        aP[e]     = pl[ka0 + e];
        aP[e + 8] = pl[8 + ka0 + e];
      }
    }
    // ---- O += P @ V  (8 column tiles of head_dim) ----
    const float* vp = cv + kvbase + (size_t)(kb + kb0) * KVROW + l16;
#pragma unroll
    for (int ct = 0; ct < 8; ++ct) {
      v16bf bV;
#pragma unroll
      for (int e = 0; e < 16; ++e)
        bV[e] = (__bf16)vp[(size_t)e * KVROW + ct * 16];
      acc[ct] = wmma_bf16(aP, bV, acc[ct]);
    }
  }

  // ---- spill per-wave partials ----
  if (l16 == 0) {
#pragma unroll
    for (int r = 0; r < 8; ++r) {
      m_lds[w][r + half * 8] = m_r[r];
      l_lds[w][r + half * 8] = l_r[r];
    }
  }
#pragma unroll
  for (int ct = 0; ct < 8; ++ct)
#pragma unroll
    for (int r = 0; r < 8; ++r)
      o_lds[w][(r + half * 8) * HD + ct * 16 + l16] = acc[ct][r];
  __syncthreads();

  // ---- cross-wave flash combine, write normalized output ----
  for (int i = threadIdx.x; i < 16 * HD; i += FA_WAVES * 32) {
    const int row = i >> 7, col = i & 127;
    float M = -INFINITY;
#pragma unroll
    for (int q = 0; q < FA_WAVES; ++q) M = fmaxf(M, m_lds[q][row]);
    float L = 0.f, O = 0.f;
#pragma unroll
    for (int q = 0; q < FA_WAVES; ++q) {
      float e = __expf(m_lds[q][row] - M);
      L += e * l_lds[q][row];
      O += e * o_lds[q][row * HD + col];
    }
    attn[(size_t)(b * SEQLEN + row) * QROW + h * HD + col] = O / L;
  }
}

// ---------------------------------------------------------------------------
// Host launcher
// ---------------------------------------------------------------------------
extern "C" void kernel_launch(void* const* d_in, const int* in_sizes, int n_in,
                              void* d_out, int out_size, void* d_ws, size_t ws_size,
                              hipStream_t stream) {
  const float* x       = (const float*)d_in[0];
  const float* mask    = (const float*)d_in[1];
  const float* freqs   = (const float*)d_in[2];
  const float* cache_k = (const float*)d_in[3];
  const float* cache_v = (const float*)d_in[4];
  const float* wq      = (const float*)d_in[5];
  const float* wk      = (const float*)d_in[6];
  const float* wv      = (const float*)d_in[7];
  const float* wo      = (const float*)d_in[8];

  // d_out = [out (128x4096) | cache_k (8x4096x1024) | cache_v (8x4096x1024)]
  float* out    = (float*)d_out;
  float* ck_out = out + (size_t)MROWS * QROW;                  // +524288
  float* cv_out = ck_out + (size_t)BSZ * CL * KVROW;           // +33554432

  // workspace layout (floats): xq | xk | xv | attn   (~5.25 MB total)
  float* ws   = (float*)d_ws;
  float* xq   = ws;                                            // 128*4096
  float* xk   = xq + (size_t)MROWS * QROW;                     // 128*1024
  float* xv   = xk + (size_t)MROWS * KVROW;                    // 128*1024
  float* attn = xv + (size_t)MROWS * KVROW;                    // 128*4096

  // 1) cache shift (bulk contiguous copy into output caches)
  shift_cache_kernel<<<2048, 256, 0, stream>>>(cache_k, cache_v, ck_out, cv_out);

  // 2) QKV projections (bf16 WMMA, f32 accumulate); 16x64 strips per wave
  gemm_bf16_kernel<<<64, 256, 0, stream>>>(x, wq, xq, MROWS, QROW, DIM);   // 512 strips
  gemm_bf16_kernel<<<16, 256, 0, stream>>>(x, wk, xk, MROWS, KVROW, DIM);  // 128 strips
  gemm_bf16_kernel<<<16, 256, 0, stream>>>(x, wv, xv, MROWS, KVROW, DIM);

  // 3) RoPE (q in place, k -> cache tail) + v -> cache tail
  rope_scatter_kernel<<<1792, 256, 0, stream>>>(xq, xk, xv, freqs, ck_out, cv_out);

  // 4) flash attention over the new caches (8 waves x 512 keys per (b,h))
  flash_attn_kernel<<<BSZ * NH, FA_WAVES * 32, 0, stream>>>(xq, ck_out, cv_out,
                                                            mask, attn);

  // 5) output projection
  gemm_bf16_kernel<<<64, 256, 0, stream>>>(attn, wo, out, MROWS, QROW, DIM);
}
